// BasicS2ConvV2_62337155334547
// MI455X (gfx1250) — matrix operand order repaired and checked by reading, not compiled, not run on hardware
//
#include <hip/hip_runtime.h>

typedef __attribute__((ext_vector_type(16))) __bf16 v16bf;
typedef __attribute__((ext_vector_type(8)))  __bf16 v8bf;
typedef __attribute__((ext_vector_type(8)))  float  v8f;
typedef __attribute__((ext_vector_type(4)))  float  v4f;
typedef __attribute__((ext_vector_type(2)))  float  v2f;
typedef __attribute__((ext_vector_type(2)))  unsigned int v2u;

#define BB    8
#define CIN   16
#define COUT  16
#define KS_   13
#define PP    4096
#define AS_   12
#define RR    12
#define KPAD  160            // per-c K (156 real + 4 zero pad) -> 5 chunks of 32
#define KSTR  168            // LDS column stride in halfs (bank spread)
#define PT    128            // p-tile per workgroup (8 waves x 16 columns)

// LDS byte layout
#define XH_OFF 0
#define XL_OFF 43008
#define AH_OFF 86016
#define AL_OFF 116736
#define LDS_BYTES 147456

#define AFRAG_HALFS_PER_RC 2560     // 5 chunks * 32 lanes * 16 halfs
#define A_PLANE_ELEMS (12*16*AFRAG_HALFS_PER_RC)   // 491520 halfs per plane

// ---------------------------------------------------------------------------
// Prep: gather Wfull[d,c,r,k,a] and store as bf16 hi/lo planes in per-lane
// WMMA A-fragment order: AH[r][c][q][lane][j] (16 halfs per lane per chunk).
// ---------------------------------------------------------------------------
__global__ void s2conv_prep_w(const float* __restrict__ W,
                              const int* __restrict__ idx_map,
                              const int* __restrict__ tivr,
                              const int* __restrict__ tir,
                              __bf16* __restrict__ AH,
                              __bf16* __restrict__ AL) {
    int idx = blockIdx.x * 256 + threadIdx.x;      // r(12) c(16) d(16) t(160)
    if (idx >= 12 * 16 * 16 * 160) return;
    int t = idx % 160;
    int d = (idx / 160) % 16;
    int c = (idx / (160 * 16)) % 16;
    int r = idx / (160 * 16 * 16);

    float val = 0.0f;
    if (t < KS_ * AS_) {
        int k = t / AS_, a = t % AS_;
        int col = idx_map[tivr[r * KS_ + k] * AS_ + tir[r * AS_ + a]];
        val = W[(d * CIN + c) * 36 + col];
    }
    __bf16 hi = (__bf16)val;
    __bf16 lo = (__bf16)(val - (float)hi);

    // ISA 16-bit A-matrix layout (16x32): lanes<16 hold K 0-7 & 16-23,
    // lanes>=16 hold K 8-15 & 24-31; element j = VGPR j/2 half j%2.
    int q  = t >> 5;
    int kk = t & 31;
    int g  = kk >> 3;                         // 0..3
    int lane = d + ((g & 1) << 4);
    int j = (g == 0) ? kk : ((g == 3) ? (kk - 16) : (kk - 8));

    long off = ((((long)r * CIN + c) * 5 + q) * 32 + lane) * 16 + j;
    AH[off] = hi;
    AL[off] = lo;
}

// ---------------------------------------------------------------------------
// Main kernel
// ---------------------------------------------------------------------------
__device__ __forceinline__ v16bf cat8(v8bf a, v8bf b) {
    return __builtin_shufflevector(a, b, 0,1,2,3,4,5,6,7,8,9,10,11,12,13,14,15);
}

__device__ __forceinline__ void split2(float v0, float v1, unsigned &h, unsigned &l) {
    __bf16 h0 = (__bf16)v0, h1 = (__bf16)v1;
    __bf16 l0 = (__bf16)(v0 - (float)h0);
    __bf16 l1 = (__bf16)(v1 - (float)h1);
    h = (unsigned)__builtin_bit_cast(unsigned short, h0) |
        ((unsigned)__builtin_bit_cast(unsigned short, h1) << 16);
    l = (unsigned)__builtin_bit_cast(unsigned short, l0) |
        ((unsigned)__builtin_bit_cast(unsigned short, l1) << 16);
}

__global__ __launch_bounds__(256) void s2conv_main(
        const float*  __restrict__ x,
        const __bf16* __restrict__ AH,
        const __bf16* __restrict__ AL,
        float*        __restrict__ out) {
    extern __shared__ char smem[];
    __bf16* XH = (__bf16*)(smem + XH_OFF);
    __bf16* XL = (__bf16*)(smem + XL_OFF);

    const int tid  = threadIdx.x;
    const int lane = tid & 31;
    const int w    = tid >> 5;                 // wave id: owns columns w*16..w*16+15
    const int b    = blockIdx.x >> 5;          // 8 batches
    const int p0   = (blockIdx.x & 31) * PT;   // 32 p-tiles

    v8f acc[RR] = {};                          // 12 rotations, f32 16x16 tiles

    // zero-pad K slots [156,160) of the x tile once (never overwritten later)
    if (tid < PT) {
        int off = tid * KSTR + KS_ * AS_;
        *(v2u*)(XH + off) = (v2u){0u, 0u};
        *(v2u*)(XL + off) = (v2u){0u, 0u};
    }

    const int ps = tid >> 1;                   // p_local this thread stages
    const int a0 = (tid & 1) * 6;              // half of the a-run

    for (int c = 0; c < CIN; ++c) {
        __syncthreads();                       // protect x tile (WAR)

        // ---- stage x[b,c,:,p0:p0+128,:] -> split bf16 hi/lo -> LDS [n][K] ----
        {
            const float* g = x + ((((long)(b * CIN + c) * KS_) * PP + (p0 + ps)) * AS_ + a0);
            for (int k = 0; k < KS_; ++k) {
                v2f f0 = __builtin_nontemporal_load((const v2f*)(g));
                v2f f1 = __builtin_nontemporal_load((const v2f*)(g + 2));
                v2f f2 = __builtin_nontemporal_load((const v2f*)(g + 4));
                unsigned h0, h1, h2, l0, l1, l2;
                split2(f0.x, f0.y, h0, l0);
                split2(f1.x, f1.y, h1, l1);
                split2(f2.x, f2.y, h2, l2);
                const int off = ps * KSTR + k * AS_ + a0;
                *(v2u*)(XH + off)            = (v2u){h0, h1};
                *((unsigned*)(XH + off) + 2) = h2;
                *(v2u*)(XL + off)            = (v2u){l0, l1};
                *((unsigned*)(XL + off) + 2) = l2;
                g += (long)PP * AS_;
            }
            if (c + 1 < CIN)  // warm L2 for the next c slab
                __builtin_prefetch(x + ((((long)(b * CIN + c + 1) * KS_) * PP + (p0 + ps)) * AS_ + a0), 0, 1);
        }

        // ---- two groups of 6 rotations; weights staged via async-to-LDS ----
        #pragma unroll
        for (int rg = 0; rg < 2; ++rg) {
            __syncthreads();                   // x ready / prior A consumers done

            for (int u = tid; u < 3840; u += 256) {       // 61440 B in b128 units
                const int plane = (u >= 1920) ? 1 : 0;
                const int v  = plane ? (u - 1920) : u;
                const int rr = v / 320;
                const int blk = v - rr * 320;
                const int r = rg * 6 + rr;
                const __bf16* srcp = (plane ? AL : AH) +
                                     (long)(r * CIN + c) * AFRAG_HALFS_PER_RC + blk * 8;
                char* dstp = smem + (plane ? AL_OFF : AH_OFF) + rr * 5120 + blk * 16;
                unsigned ldsaddr = (unsigned)(size_t)dstp;
                asm volatile("global_load_async_to_lds_b128 %0, %1, off"
                             :: "v"(ldsaddr), "v"(srcp) : "memory");
            }
            asm volatile("s_wait_asynccnt 0" ::: "memory");
            __syncthreads();                   // A fragments visible to all waves

            // ---- WMMA over 5 K-chunks of 32 ----
            const int coloff = (w * 16 + (lane & 15)) * KSTR + ((lane >> 4) << 3);
            for (int q = 0; q < 5; ++q) {
                const int bo = coloff + q * 32;
                v8bf bh0 = *(const v8bf*)(XH + bo);
                v8bf bh1 = *(const v8bf*)(XH + bo + 16);
                v8bf bl0 = *(const v8bf*)(XL + bo);
                v8bf bl1 = *(const v8bf*)(XL + bo + 16);
                v16bf bh = cat8(bh0, bh1);
                v16bf bl = cat8(bl0, bl1);
                #pragma unroll
                for (int rr = 0; rr < 6; ++rr) {
                    const char* ahp = smem + AH_OFF + (rr * 5 + q) * 1024 + lane * 32;
                    const char* alp = smem + AL_OFF + (rr * 5 + q) * 1024 + lane * 32;
                    v16bf ah = cat8(*(const v8bf*)ahp, *(const v8bf*)(ahp + 16));
                    v16bf al = cat8(*(const v8bf*)alp, *(const v8bf*)(alp + 16));
                    const int r = rg * 6 + rr;
                    acc[r] = __builtin_amdgcn_wmma_f32_16x16x32_bf16(
                                 false, al, false, bh, (short)0, acc[r], false, false);
                    acc[r] = __builtin_amdgcn_wmma_f32_16x16x32_bf16(
                                 false, ah, false, bl, (short)0, acc[r], false, false);
                    acc[r] = __builtin_amdgcn_wmma_f32_16x16x32_bf16(
                                 false, ah, false, bh, (short)0, acc[r], false, false);
                }
            }
        }
    }

    // ---- write out[b,d,p,r]: 12 contiguous r-values per (d,p) -> 3x b128 ----
    const int d0 = (lane >> 4) * 8;            // C-layout: VGPR i -> M = d0 + i
    const int pp = p0 + w * 16 + (lane & 15);
    #pragma unroll
    for (int i = 0; i < 8; ++i) {
        const int d = d0 + i;
        float* o = out + ((long)(b * COUT + d) * PP + pp) * RR;
        v4f s0 = {acc[0][i], acc[1][i], acc[2][i],  acc[3][i]};
        v4f s1 = {acc[4][i], acc[5][i], acc[6][i],  acc[7][i]};
        v4f s2 = {acc[8][i], acc[9][i], acc[10][i], acc[11][i]};
        __builtin_nontemporal_store(s0, (v4f*)(o));
        __builtin_nontemporal_store(s1, (v4f*)(o + 4));
        __builtin_nontemporal_store(s2, (v4f*)(o + 8));
    }
}

// ---------------------------------------------------------------------------
extern "C" void kernel_launch(void* const* d_in, const int* in_sizes, int n_in,
                              void* d_out, int out_size, void* d_ws, size_t ws_size,
                              hipStream_t stream) {
    const float* x       = (const float*)d_in[0];
    const float* W       = (const float*)d_in[1];
    const int*   idx_map = (const int*)d_in[2];
    const int*   tivr    = (const int*)d_in[3];
    const int*   tir     = (const int*)d_in[4];

    __bf16* AH = (__bf16*)d_ws;
    __bf16* AL = AH + A_PLANE_ELEMS;

    s2conv_prep_w<<<1920, 256, 0, stream>>>(W, idx_map, tivr, tir, AH, AL);
    s2conv_main<<<BB * (PP / PT), 256, LDS_BYTES, stream>>>(x, AH, AL, (float*)d_out);
}
